// StateQuantizer_2121713844444
// MI455X (gfx1250) — matrix operand order repaired and checked by reading, not compiled, not run on hardware
//
#include <hip/hip_runtime.h>

typedef __attribute__((ext_vector_type(16))) __bf16 v16bf;
typedef __attribute__((ext_vector_type(16))) unsigned short us16;
typedef __attribute__((ext_vector_type(8)))  float v8f;
typedef __attribute__((ext_vector_type(4)))  unsigned int u32x4;
typedef __attribute__((ext_vector_type(8)))  int i32x8;
typedef __attribute__((ext_vector_type(4)))  int i32x4;

#define D_DIM 64
#define K_CODES 1024
#define CHUNK 128            // codes per LDS buffer (16 KB bf16, double-buffered)
#define NCHUNKS (K_CODES / CHUNK)
#define NTILES (CHUNK / 16)  // 16-code N-tiles per chunk
#define THREADS 256          // 8 waves (wave32)
#define ROWS_PER_BLOCK 128   // 16 rows per wave
#define M_ROWS 65536

#if defined(__AMDGCN__) && __has_builtin(__builtin_amdgcn_tensor_load_to_lds)
#define HAVE_TDM 1
#else
#define HAVE_TDM 0
#endif

__device__ __forceinline__ unsigned short f2bf(float x) {
  unsigned int u = __builtin_bit_cast(unsigned int, x);
  u += 0x7FFFu + ((u >> 16) & 1u);      // round to nearest even
  return (unsigned short)(u >> 16);
}

// ---- prep: codebook f32 -> bf16, plus per-code squared norms ----
__global__ void vq_prep(const float* __restrict__ cb,
                        unsigned short* __restrict__ cb_bf,
                        float* __restrict__ cn) {
  int c = blockIdx.x * blockDim.x + threadIdx.x;
  if (c < K_CODES) {
    float s = 0.f;
    #pragma unroll
    for (int d = 0; d < D_DIM; ++d) {
      float v = cb[c * D_DIM + d];
      s += v * v;
      cb_bf[c * D_DIM + d] = f2bf(v);
    }
    cn[c] = s;
  }
}

// ---- fused distance-GEMM + argmin + gather + loss partials ----
__global__ __launch_bounds__(THREADS) void vq_main(
    const float* __restrict__ z,
    const float* __restrict__ cb,
    const unsigned short* __restrict__ cb_bf,
    const float* __restrict__ cn,
    float* __restrict__ out,
    float* __restrict__ partials) {
  __shared__ unsigned short lds_cb[2][CHUNK * D_DIM];  // 2 x 16 KB
  __shared__ float lds_cn[K_CODES];                    // 4 KB (all norms)
  __shared__ float lds_red[THREADS / 32];

  const int lane = threadIdx.x & 31;
  const int wave = threadIdx.x >> 5;
  const int hh   = lane >> 4;     // half-wave id
  const int l15  = lane & 15;
  const int rowBase = blockIdx.x * ROWS_PER_BLOCK + wave * 16;

  // stage all code norms once
  #pragma unroll
  for (int i = 0; i < K_CODES / THREADS; ++i)
    lds_cn[threadIdx.x + i * THREADS] = cn[threadIdx.x + i * THREADS];

#if HAVE_TDM
  // Tensor Data Mover: one descriptor DMA per 16 KB codebook chunk,
  // issued by wave 0, tracked on TENSORcnt, overlapped with compute.
  auto tdm_issue = [&](int chunk, int buf) {
    unsigned long long ga =
        (unsigned long long)(size_t)(cb_bf + (size_t)chunk * CHUNK * D_DIM);
    unsigned lds_addr = (unsigned)(size_t)&lds_cb[buf][0];
    // D# group 0: count=1 | lds_addr | global_addr[56:0] | type=2
    u32x4 g0;
    g0[0] = 1u;                                   // count=1, user mode
    g0[1] = lds_addr;                             // [63:32] lds_addr
    g0[2] = (unsigned)(ga & 0xFFFFFFFFu);         // [95:64] global_addr lo
    g0[3] = (unsigned)((ga >> 32) & 0x01FFFFFFu)  // [120:96] global_addr hi
          | (2u << 30);                           // [127:126] type=2 (image)
    // D# group 1: data_size=8B; 1-D tile: tensor_dim0=tile_dim0=2048 elems
    i32x8 g1;
    g1[0] = (int)(3u << 16);        // workgroup_mask=0, data_size=3 (8B)
    g1[1] = (int)(2048u << 16);     // tensor_dim0[15:0] @ bits 63:48
    g1[2] = (int)(1u << 16);        // tensor_dim0 hi=0, tensor_dim1=1
    g1[3] = (int)(2048u << 16);     // tensor_dim1 hi=0, tile_dim0=2048
    g1[4] = 1;                      // tile_dim1=1, tile_dim2=0
    g1[5] = 2048;                   // tensor_dim0_stride lo32
    g1[6] = 0;                      // stride0 hi, tensor_dim1_stride lo
    g1[7] = 0;
    i32x4 gz4 = {0, 0, 0, 0};           // groups 2/3 unused (2-D tensor)
    i32x8 gz8 = {0, 0, 0, 0, 0, 0, 0, 0};
    // amdgpu-toolchain (clang-23) 6-arg form
    __builtin_amdgcn_tensor_load_to_lds(g0, g1, gz4, gz4, gz8, 0);
  };
#else
  auto stage_sync = [&](int chunk, int buf) {
    const uint4* src = (const uint4*)(cb_bf + (size_t)chunk * CHUNK * D_DIM);
    uint4* dst = (uint4*)&lds_cb[buf][0];
    #pragma unroll
    for (int i = 0; i < (CHUNK * D_DIM / 8) / THREADS; ++i)   // 1024 uint4
      dst[threadIdx.x + i * THREADS] = src[threadIdx.x + i * THREADS];
    if (chunk + 1 < NCHUNKS)
      __builtin_prefetch(cb_bf + (size_t)(chunk + 1) * CHUNK * D_DIM, 0, 0);
  };
#endif

  // A fragments (16x32 bf16, contiguous k order, same as B):
  // lane(hh,l15): M = l15; elem j: K = 16*hh + j
  v16bf a[2];
  {
    const float* zrow = z + (size_t)(rowBase + l15) * D_DIM;
    #pragma unroll
    for (int f = 0; f < 2; ++f) {
      us16 t;
      const int d0 = f * 32 + 16 * hh;
      #pragma unroll
      for (int j = 0; j < 8; ++j) t[j]     = f2bf(zrow[d0 + j]);
      #pragma unroll
      for (int j = 0; j < 8; ++j) t[8 + j] = f2bf(zrow[d0 + 8 + j]);
      a[f] = __builtin_bit_cast(v16bf, t);
    }
  }

  float bestd[8];
  int   bestn[8];
  #pragma unroll
  for (int r = 0; r < 8; ++r) { bestd[r] = 3.4e38f; bestn[r] = 0; }

  // prolog: chunk 0 into buffer 0
#if HAVE_TDM
  if (wave == 0) {
    tdm_issue(0, 0);
    __builtin_amdgcn_s_wait_tensorcnt(0);
  }
#else
  stage_sync(0, 0);
#endif
  __syncthreads();

  for (int c = 0; c < NCHUNKS; ++c) {
    const int cur = c & 1;
#if HAVE_TDM
    if (c + 1 < NCHUNKS && wave == 0) tdm_issue(c + 1, cur ^ 1);  // overlap DMA
#else
    if (c + 1 < NCHUNKS) stage_sync(c + 1, cur ^ 1);
#endif

    const int chunkBase = c * CHUNK;
    // B fragments (32x16 bf16): lane(hh,l15): N = l15; elem j: K = 16*hh + j
    // lane's base: 32 contiguous bytes at row (n0+l15), col offset 16*hh;
    // tile stride = 16 codes * 64 elems = 64 us16 units.
    const us16* bbase = (const us16*)&lds_cb[cur][l15 * D_DIM + 16 * hh];

    // software pipeline: B fragments for tile t+1 load during tile t compute
    v16bf b0 = __builtin_bit_cast(v16bf, bbase[0]);
    v16bf b1 = __builtin_bit_cast(v16bf, bbase[2]);
    #pragma unroll 2
    for (int t = 0; t < NTILES; ++t) {
      v16bf nb0 = b0, nb1 = b1;
      if (t + 1 < NTILES) {
        const us16* np = bbase + (t + 1) * 64;
        nb0 = __builtin_bit_cast(v16bf, np[0]);
        nb1 = __builtin_bit_cast(v16bf, np[2]);
      }

      v8f acc = {};
      acc = __builtin_amdgcn_wmma_f32_16x16x32_bf16(false, a[0], false, b0,
                                                    (short)0, acc, false, false);
      acc = __builtin_amdgcn_wmma_f32_16x16x32_bf16(false, a[1], false, b1,
                                                    (short)0, acc, false, false);

      // dist = ||c_n||^2 - 2 z.c_n  (||z||^2 constant per row, dropped)
      const int n = chunkBase + t * 16 + l15;
      const float cnv = lds_cn[n];
      #pragma unroll
      for (int r = 0; r < 8; ++r) {
        float d = __builtin_fmaf(-2.0f, acc[r], cnv);
        if (d < bestd[r]) { bestd[r] = d; bestn[r] = n; }
      }
      b0 = nb0; b1 = nb1;
    }

#if HAVE_TDM
    if (wave == 0) __builtin_amdgcn_s_wait_tensorcnt(0);
#endif
    __syncthreads();
  }

  // cross-lane argmin within each 16-lane half (first-index tie-break)
  #pragma unroll
  for (int off = 8; off >= 1; off >>= 1) {
    #pragma unroll
    for (int r = 0; r < 8; ++r) {
      float dd = __shfl_xor(bestd[r], off, 32);
      int   nn = __shfl_xor(bestn[r], off, 32);
      if (dd < bestd[r] || (dd == bestd[r] && nn < bestn[r])) {
        bestd[r] = dd; bestn[r] = nn;
      }
    }
  }

  // gather exact f32 codes, write z_q_st, accumulate loss partial
  float lsum = 0.f;
  #pragma unroll
  for (int r = 0; r < 8; ++r) {
    const int row = rowBase + 8 * hh + r;     // C/D layout: M = r + 8*hh
    const int idx = bestn[r];
    const int c0  = l15 * 4;
    const float4 q  = *(const float4*)(cb + (size_t)idx * D_DIM + c0);
    const float4 zv = *(const float4*)(z  + (size_t)row * D_DIM + c0);
    *(float4*)(out + (size_t)row * D_DIM + c0) = q;
    float dx = q.x - zv.x, dy = q.y - zv.y, dz = q.z - zv.z, dw = q.w - zv.w;
    lsum += dx * dx + dy * dy + dz * dz + dw * dw;
  }

  #pragma unroll
  for (int off = 16; off >= 1; off >>= 1) lsum += __shfl_xor(lsum, off, 32);
  if (lane == 0) lds_red[wave] = lsum;
  __syncthreads();
  if (threadIdx.x == 0) {
    float s = 0.f;
    #pragma unroll
    for (int w = 0; w < THREADS / 32; ++w) s += lds_red[w];
    partials[blockIdx.x] = s;
  }
}

// ---- deterministic final loss reduction over 512 block partials ----
__global__ void vq_loss(const float* __restrict__ part,
                        float* __restrict__ loss) {
  __shared__ float s[512];
  s[threadIdx.x] = part[threadIdx.x];
  __syncthreads();
  for (int off = 256; off >= 1; off >>= 1) {
    if ((int)threadIdx.x < off) s[threadIdx.x] += s[threadIdx.x + off];
    __syncthreads();
  }
  if (threadIdx.x == 0)
    *loss = 11.0f * s[0] / (float)(M_ROWS * D_DIM);   // (0.1+1)*10 * mse
}

extern "C" void kernel_launch(void* const* d_in, const int* in_sizes, int n_in,
                              void* d_out, int out_size, void* d_ws, size_t ws_size,
                              hipStream_t stream) {
  const float* z  = (const float*)d_in[0];   // (32,2048,64) f32
  const float* cb = (const float*)d_in[1];   // (1024,64) f32
  float* out = (float*)d_out;                // 4194304 z_q + 1 loss

  unsigned short* cb_bf = (unsigned short*)d_ws;                       // 128 KB
  float* cn       = (float*)((char*)d_ws + K_CODES * D_DIM * 2);       // 4 KB
  float* partials = (float*)((char*)d_ws + K_CODES * D_DIM * 2
                                         + K_CODES * 4);               // 2 KB

  vq_prep<<<(K_CODES + 255) / 256, 256, 0, stream>>>(cb, cb_bf, cn);
  vq_main<<<M_ROWS / ROWS_PER_BLOCK, THREADS, 0, stream>>>(
      z, cb, cb_bf, cn, out, partials);
  vq_loss<<<1, 512, 0, stream>>>(partials, out + (size_t)M_ROWS * D_DIM);
}
